// SimpleTransformer_36850819400039
// MI455X (gfx1250) — matrix operand order repaired and checked by reading, not compiled, not run on hardware
//
#include <hip/hip_runtime.h>
#include <hip/hip_bf16.h>

#define BATCH 32
#define CCH   64
#define NTOK  1024
#define DEPTH 8
#define NCLS  10

typedef __attribute__((ext_vector_type(16))) __bf16 v16bf;
typedef __attribute__((ext_vector_type(8)))  float  v8f;

__device__ __forceinline__ unsigned short f2bf(float f) {
  union { float f; unsigned u; } v; v.f = f;
  unsigned r = v.u + 0x7FFFu + ((v.u >> 16) & 1u);   // round-to-nearest-even
  return (unsigned short)(r >> 16);
}

// Convert two f32 -> packed bf16x2 (low = a, high = b).
__device__ __forceinline__ unsigned f2bf2(float a, float b) {
#if __has_builtin(__builtin_amdgcn_cvt_pk_bf16_f32)
  typedef __attribute__((ext_vector_type(2))) __bf16 v2bf;
  union { v2bf v; unsigned u; } u;
  u.v = __builtin_amdgcn_cvt_pk_bf16_f32(a, b);
  return u.u;
#else
  return (unsigned)f2bf(a) | ((unsigned)f2bf(b) << 16);
#endif
}

union PairBF { unsigned u; unsigned short s[2]; };

union FragBF {
  v16bf v;
  unsigned short s[16];
  unsigned w[8];
  uint4 q[2];
};

// A fragment, 16x32 bf16 (MxK), K-stride == 1 -> two contiguous 16B loads.
__device__ __forceinline__ v16bf load_a_k1(const unsigned short* base, int sm, int lane) {
  int m = lane & 15, hi = lane >> 4;
  const unsigned short* p = base + m * sm + (hi << 3);
  FragBF u;
  u.q[0] = *(const uint4*)p;
  u.q[1] = *(const uint4*)(p + 16);
  return u.v;
}

// B fragment, 32x16 bf16 (KxN), K-stride == 1 -> two contiguous 16B loads.
__device__ __forceinline__ v16bf load_b_k1(const unsigned short* base, int sn, int lane) {
  int n = lane & 15, hi = lane >> 4;
  const unsigned short* p = base + n * sn + (hi << 3);
  FragBF u;
  u.q[0] = *(const uint4*)p;
  u.q[1] = *(const uint4*)(p + 16);
  return u.v;
}

// ---- weight conversion: WQ,WK,WV (f32, [k][n]) -> bf16 transposed Wt[which][n][k]
__global__ __launch_bounds__(256) void wconv_kernel(
    const float* __restrict__ WQ, const float* __restrict__ WK,
    const float* __restrict__ WV, unsigned short* __restrict__ Wt) {
  int idx = blockIdx.x * 256 + threadIdx.x;          // 0 .. 3*4096-1
  int which = idx >> 12;
  int rem   = idx & 4095;
  int n = rem >> 6, k = rem & 63;
  const float* src = (which == 0) ? WQ : (which == 1) ? WK : WV;
  Wt[idx] = f2bf(src[k * 64 + n]);
}

// ---- init: image (B,C,H,W) -> X (B, NTOK, C) f32 + bf16 --------------------
__global__ __launch_bounds__(256) void init_kernel(
    const float* __restrict__ img, float* __restrict__ Xf,
    unsigned short* __restrict__ Xbf) {
  int idx = blockIdx.x * 256 + threadIdx.x;
  int c   = idx & 63;
  int rem = idx >> 6;
  int d   = rem & 1023;
  int b   = rem >> 10;
  float v = img[(b * CCH + c) * NTOK + d];
  Xf[idx]  = v;
  Xbf[idx] = f2bf(v);
}

// ---- QKV: per-wave 16x16 output tile of X @ W ------------------------------
// Q row-major PRE-SCALED by 1/32; K row-major; V TRANSPOSED (Vt[b][ch][token]).
__global__ __launch_bounds__(256) void qkv_kernel(
    const unsigned short* __restrict__ Xbf, const unsigned short* __restrict__ Wt,
    unsigned short* __restrict__ Qbf, unsigned short* __restrict__ Kbf,
    unsigned short* __restrict__ Vt) {
  int tid  = threadIdx.x;
  int lane = tid & 31;
  int wave = tid >> 5;
  int t    = blockIdx.x * 8 + wave;
  int col  = t & 3;  t >>= 2;
  int row  = t & 63; t >>= 6;
  int out  = t % 3;
  int b    = t / 3;

  const unsigned short* A  = Xbf + (size_t)b * (NTOK * CCH) + row * 16 * CCH;
  const unsigned short* Bw = Wt + out * 4096 + (col * 16) * 64;   // Wt[n][k]

  v8f acc = {};
#pragma unroll
  for (int k0 = 0; k0 < CCH; k0 += 32) {
    v16bf a  = load_a_k1(A + k0, CCH, lane);
    v16bf bb = load_b_k1(Bw + k0, 64, lane);
    acc = __builtin_amdgcn_wmma_f32_16x16x32_bf16(false, a, false, bb,
                                                  (short)0, acc, false, false);
  }
  int n = lane & 15, hi = lane >> 4;
  if (out == 2) {
    // Vt[b][col*16+n][row*16 + 8*hi + r] : 8 consecutive tokens -> 1x 16B store
    FragBF pk;
#pragma unroll
    for (int r = 0; r < 8; r += 2) pk.w[r >> 1] = f2bf2(acc[r], acc[r + 1]);
    unsigned short* O = Vt + (size_t)b * (NTOK * CCH)
                      + (col * 16 + n) * NTOK + row * 16 + 8 * hi;
    *(uint4*)O = pk.q[0];
  } else {
    float sc = (out == 0) ? 0.03125f : 1.0f;   // fold softmax scale into Q
    unsigned short* O = (out == 0) ? Qbf : Kbf;
    O += (size_t)b * (NTOK * CCH) + (row * 16 + 8 * hi) * CCH + col * 16 + n;
#pragma unroll
    for (int r = 0; r < 8; r += 2) {
      PairBF pk; pk.u = f2bf2(acc[r] * sc, acc[r + 1] * sc);
      O[r * CCH]       = pk.s[0];
      O[(r + 1) * CCH] = pk.s[1];
    }
  }
}

// ---- attention: one block = (batch, 16 query rows), 8 waves ---------------
__global__ __launch_bounds__(256) void attn_kernel(
    const unsigned short* __restrict__ Qbf, const unsigned short* __restrict__ Kbf,
    const unsigned short* __restrict__ Vt, const float* __restrict__ Xin,
    float* __restrict__ Xout, unsigned short* __restrict__ Xbf_out) {
  __shared__ alignas(16) unsigned short Pl[16 * NTOK];  // unnormalized probs (32KB)
  __shared__ float wred[8][16];
  __shared__ float rowMax[16];
  __shared__ float rowInv[16];
  __shared__ float opart[4][16][16];

  int tid  = threadIdx.x;
  int lane = tid & 31;
  int wave = tid >> 5;
  int hi   = lane >> 4;
  int nl   = lane & 15;
  int b    = blockIdx.x >> 6;
  int row0 = (blockIdx.x & 63) * 16;

  const unsigned short* Qb = Qbf + (size_t)b * (NTOK * CCH) + row0 * CCH;
  const unsigned short* Kb = Kbf + (size_t)b * (NTOK * CCH);
  const unsigned short* Vb = Vt  + (size_t)b * (NTOK * CCH);

  // ---- Phase A: S = Qs @ K^T (Q pre-scaled). Q fragments hoisted. ----
  v16bf qa0 = load_a_k1(Qb +  0, CCH, lane);
  v16bf qa1 = load_a_k1(Qb + 32, CCH, lane);
  float st[8][8];
#pragma unroll
  for (int t = 0; t < 8; ++t) {
    int n0 = wave * 128 + t * 16;
    v8f acc = {};
    v16bf kb0 = load_b_k1(Kb + n0 * CCH +  0, CCH, lane);
    v16bf kb1 = load_b_k1(Kb + n0 * CCH + 32, CCH, lane);
    acc = __builtin_amdgcn_wmma_f32_16x16x32_bf16(false, qa0, false, kb0,
                                                  (short)0, acc, false, false);
    acc = __builtin_amdgcn_wmma_f32_16x16x32_bf16(false, qa1, false, kb1,
                                                  (short)0, acc, false, false);
#pragma unroll
    for (int r = 0; r < 8; ++r) st[t][r] = acc[r];
  }

  // ---- row max: per-thread, then across the 16 lanes of the half-wave ----
#pragma unroll
  for (int r = 0; r < 8; ++r) {
    float m = st[0][r];
#pragma unroll
    for (int t = 1; t < 8; ++t) m = fmaxf(m, st[t][r]);
#pragma unroll
    for (int off = 1; off < 16; off <<= 1) m = fmaxf(m, __shfl_xor(m, off, 32));
    if (nl == 0) wred[wave][hi * 8 + r] = m;
  }
  __syncthreads();
  if (tid < 16) {
    float m = wred[0][tid];
#pragma unroll
    for (int w = 1; w < 8; ++w) m = fmaxf(m, wred[w][tid]);
    rowMax[tid] = m;
  }
  __syncthreads();

  // ---- Phase B: P = exp(S - max) -> bf16 LDS (paired cvt); row sums ----
  float vsum[8];
#pragma unroll
  for (int r = 0; r < 8; ++r) vsum[r] = 0.0f;
  float mx[8];
#pragma unroll
  for (int r = 0; r < 8; ++r) mx[r] = rowMax[r + 8 * hi];
#pragma unroll
  for (int t = 0; t < 8; ++t) {
    int col = wave * 128 + t * 16 + nl;
#pragma unroll
    for (int r = 0; r < 8; r += 2) {
      float p0 = __expf(st[t][r]     - mx[r]);
      float p1 = __expf(st[t][r + 1] - mx[r + 1]);
      vsum[r]     += p0;
      vsum[r + 1] += p1;
      PairBF pk; pk.u = f2bf2(p0, p1);
      Pl[(r     + 8 * hi) * NTOK + col] = pk.s[0];
      Pl[(r + 1 + 8 * hi) * NTOK + col] = pk.s[1];
    }
  }
#pragma unroll
  for (int r = 0; r < 8; ++r) {
    float s = vsum[r];
#pragma unroll
    for (int off = 1; off < 16; off <<= 1) s += __shfl_xor(s, off, 32);
    if (nl == 0) wred[wave][hi * 8 + r] = s;
  }
  __syncthreads();
  if (tid < 16) {
    float s = wred[0][tid];
#pragma unroll
    for (int w = 1; w < 8; ++w) s += wred[w][tid];
    rowInv[tid] = 1.0f / s;
  }
  __syncthreads();

  // ---- Phase C: O = P @ V ; split-K across wave pairs; combine; residual ----
  int ct   = wave & 3;        // channel tile (4 x 16 = 64 channels)
  int half = wave >> 2;       // split-K half
  v8f oacc = {};
  const unsigned short* VtB = Vb + (ct * 16) * NTOK;  // Vt[channel][token]
#pragma unroll
  for (int kk = 0; kk < 16; ++kk) {
    int k0 = half * 512 + kk * 32;
    v16bf a  = load_a_k1(Pl + k0, NTOK, lane);     // 2x ds_load_b128
    v16bf bb = load_b_k1(VtB + k0, NTOK, lane);    // contiguous in token
    oacc = __builtin_amdgcn_wmma_f32_16x16x32_bf16(false, a, false, bb,
                                                   (short)0, oacc, false, false);
  }
  if (half == 1) {
#pragma unroll
    for (int r = 0; r < 8; ++r) opart[ct][r + 8 * hi][nl] = oacc[r];
  }
  __syncthreads();
  if (half == 0) {
    size_t off = (size_t)b * (NTOK * CCH) + (row0 + 8 * hi) * CCH + ct * 16 + nl;
    const float* Xi = Xin + off;
    float* Xo = Xout + off;
    unsigned short* Xob = Xbf_out + off;
    float xn[8];
#pragma unroll
    for (int r = 0; r < 8; ++r) {
      int m = r + 8 * hi;
      float o = (oacc[r] + opart[ct][m][nl]) * rowInv[m];
      xn[r] = o + Xi[r * CCH];                     // tau = 1.0
      Xo[r * CCH] = xn[r];
    }
#pragma unroll
    for (int r = 0; r < 8; r += 2) {
      PairBF pk; pk.u = f2bf2(xn[r], xn[r + 1]);
      Xob[r * CCH]       = pk.s[0];
      Xob[(r + 1) * CCH] = pk.s[1];
    }
  }
}

// ---- head: pred = mean_c(X) @ lastW^T + lastB ------------------------------
__global__ __launch_bounds__(256) void head_kernel(
    const float* __restrict__ X, const float* __restrict__ lastW,
    const float* __restrict__ lastB, float* __restrict__ out) {
  __shared__ float mean[NTOK];
  __shared__ float red[256];
  int b   = blockIdx.x;
  int tid = threadIdx.x;
  const float* Xb = X + (size_t)b * (NTOK * CCH);
  for (int d = tid; d < NTOK; d += 256) {
    float s = 0.0f;
    for (int c = 0; c < CCH; ++c) s += Xb[d * CCH + c];
    mean[d] = s * (1.0f / CCH);
  }
  __syncthreads();
  for (int cls = 0; cls < NCLS; ++cls) {
    float p = 0.0f;
    for (int d = tid; d < NTOK; d += 256) p += mean[d] * lastW[cls * NTOK + d];
    red[tid] = p;
    __syncthreads();
    for (int s = 128; s > 0; s >>= 1) {
      if (tid < s) red[tid] += red[tid + s];
      __syncthreads();
    }
    if (tid == 0) out[b * NCLS + cls] = red[0] + lastB[cls];
    __syncthreads();
  }
}

extern "C" void kernel_launch(void* const* d_in, const int* in_sizes, int n_in,
                              void* d_out, int out_size, void* d_ws, size_t ws_size,
                              hipStream_t stream) {
  (void)in_sizes; (void)n_in; (void)out_size; (void)ws_size;
  const float* img   = (const float*)d_in[0];
  const float* WV    = (const float*)d_in[1];
  const float* WK    = (const float*)d_in[2];
  const float* WQ    = (const float*)d_in[3];
  const float* lastW = (const float*)d_in[4];
  const float* lastB = (const float*)d_in[5];
  float* out = (float*)d_out;

  // workspace layout (~32 MB)
  char* ws = (char*)d_ws;
  const size_t XN = (size_t)BATCH * NTOK * CCH;
  float*          Xf0 = (float*)ws;          ws += XN * sizeof(float);
  float*          Xf1 = (float*)ws;          ws += XN * sizeof(float);
  unsigned short* Xbf = (unsigned short*)ws; ws += XN * sizeof(unsigned short);
  unsigned short* Qb  = (unsigned short*)ws; ws += XN * sizeof(unsigned short);
  unsigned short* Kb  = (unsigned short*)ws; ws += XN * sizeof(unsigned short);
  unsigned short* Vt  = (unsigned short*)ws; ws += XN * sizeof(unsigned short);
  unsigned short* Wt  = (unsigned short*)ws; ws += 3 * 4096 * sizeof(unsigned short);

  wconv_kernel<<<48, 256, 0, stream>>>(WQ, WK, WV, Wt);
  init_kernel<<<(int)(XN / 256), 256, 0, stream>>>(img, Xf0, Xbf);

  float* cur = Xf0;
  float* nxt = Xf1;
  for (int l = 0; l < DEPTH; ++l) {
    qkv_kernel<<<3072, 256, 0, stream>>>(Xbf, Wt, Qb, Kb, Vt);
    attn_kernel<<<2048, 256, 0, stream>>>(Qb, Kb, Vt, cur, nxt, Xbf);
    float* t = cur; cur = nxt; nxt = t;
  }
  head_kernel<<<BATCH, 256, 0, stream>>>(cur, lastW, lastB, out);
}
